// UMamba_83743272338012
// MI455X (gfx1250) — matrix operand order repaired and checked by reading, not compiled, not run on hardware
//
#include <hip/hip_runtime.h>
#include <math.h>

// ---------------------------------------------------------------------------
// U-Mamba forward for MI455X (gfx1250, wave32).
// GEMMs run on v_wmma_f32_16x16x32_f16 (f16 inputs, f32 accumulate).
// Selective scan: fp32 VALU, 16 lanes per channel (one per SSM state).
// ---------------------------------------------------------------------------

typedef __attribute__((ext_vector_type(16))) _Float16 v16h;
typedef __attribute__((ext_vector_type(8)))  _Float16 v8h;
typedef __attribute__((ext_vector_type(8)))  float    v8f;

#define DIV_UP(a,b) (((a)+(b)-1)/(b))

// ---- workspace layout (float units) ---------------------------------------
enum : size_t {
  OFF_BUFA   = 0,                       // 2,097,152  conv outputs
  OFF_BUFB   = OFF_BUFA   + 2097152,    // 2,097,152  bn/relu outputs
  OFF_X1     = OFF_BUFB   + 2097152,    // 2,097,152  x1 (skip connection)
  OFF_BIG    = OFF_X1     + 2097152,    // 8,388,608  xr (mamba) / concat
  OFF_XCONV  = OFF_BIG    + 8388608,    // 4,194,304  depthwise conv out f32
  OFF_XCONVH = OFF_XCONV  + 4194304,    // 2,097,152  (f16 copy)
  OFF_XDBL   = OFF_XCONVH + 2097152,    // 2,359,296  x_dbl f32
  OFF_DTINH  = OFF_XDBL   + 2359296,    // 1,048,576  delta-input f16 (K pad 32)
  OFF_DELTA  = OFF_DTINH  + 1048576,    // 4,194,304  softplus(delta) f32
  OFF_YSCAN  = OFF_DELTA  + 4194304,    // 4,194,304  scan output f32
  OFF_Y2H    = OFF_YSCAN  + 4194304,    // 2,097,152  gated y f16
  OFF_SEQH   = OFF_Y2H    + 2097152,    // 1,048,576  seq input f16
  OFF_MOUT   = OFF_SEQH   + 1048576,    // 2,097,152  mamba out f32 (B*L, dm)
  OFF_P1     = OFF_MOUT   + 2097152,    //   524,288  maxpool out
  OFF_X2     = OFF_P1     + 524288,     // 1,048,576  x2
  OFF_XB     = OFF_X2     + 1048576,    // 1,048,576  bottleneck mamba out
  OFF_STATS  = OFF_XB     + 1048576,    //       512  BN mean/rstd
  OFF_WH     = OFF_STATS  + 512,        //    32,768  f16 weight pool (65536 h)
};

// f16 weight pool sub-offsets (halves)
enum : size_t { WH_WIN = 0, WH_WX = 16384, WH_WDT = 24576, WH_WOUT = 28672 };

__device__ __forceinline__ float silu_f(float x)     { return x / (1.f + __expf(-x)); }
__device__ __forceinline__ float softplus_f(float x) { return x > 20.f ? x : log1pf(__expf(x)); }

// ---- f32 -> f16 weight cast with zero padding (rows x cols -> drows x dcols)
__global__ void k_cast_w16(const float* __restrict__ src, int srows, int scols,
                           _Float16* __restrict__ dst, int drows, int dcols)
{
  int idx = blockIdx.x * blockDim.x + threadIdx.x;
  if (idx >= drows * dcols) return;
  int c = idx % dcols, r = idx / dcols;
  float v = (r < srows && c < scols) ? src[(size_t)r * scols + c] : 0.f;
  dst[idx] = (_Float16)v;
}

// ---- f32 strided -> f16 padded activation cast ----------------------------
__global__ void k_cast_a16(const float* __restrict__ src, int src_ld, int ncols,
                           _Float16* __restrict__ dst, int dst_ld, int rows)
{
  int idx = blockIdx.x * blockDim.x + threadIdx.x;
  if (idx >= rows * dst_ld) return;
  int c = idx % dst_ld, r = idx / dst_ld;
  dst[idx] = (_Float16)((c < ncols) ? src[(size_t)r * src_ld + c] : 0.f);
}

// ---- NCHW -> (B*L, C) f16 -------------------------------------------------
__global__ void k_nchw_to_seq(const float* __restrict__ x, _Float16* __restrict__ seq,
                              int B, int C, int L)
{
  int idx = blockIdx.x * blockDim.x + threadIdx.x;
  if (idx >= B * C * L) return;
  int c = idx % C; int row = idx / C;
  int l = row % L; int b = row / L;
  seq[idx] = (_Float16)x[((size_t)(b * C + c)) * L + l];
}

// ---- WMMA GEMM: C[M,N] = A[M,K] * W[N,K]^T (+bias, +act) ------------------
// A: f16 row-major, lda multiple of 32. W: f16 row-major, rows padded to 16*gridDim.y.
// 4 waves / block, each wave one 16x16 tile; K stepped by 32 via v_wmma_f32_16x16x32_f16.
__global__ void k_gemm_wmma(const _Float16* __restrict__ A, int lda,
                            const _Float16* __restrict__ Wt, int ldw,
                            const float* __restrict__ bias,
                            float* __restrict__ C, int ldc, int N,
                            int ksteps, int act)
{
  const int lane  = threadIdx.x & 31;
  const int mtile = blockIdx.x * 4 + (threadIdx.x >> 5);
  const int ntile = blockIdx.y;
  const int lrow  = lane & 15;
  const int lhi   = lane >> 4;

  const _Float16* ap = A  + (size_t)(mtile * 16 + lrow) * lda + lhi * 8;
  const _Float16* wp = Wt + (size_t)(ntile * 16 + lrow) * ldw + lhi * 8;

  v8f acc = {};
  for (int ks = 0; ks < ksteps; ++ks) {
    v8h a0 = *(const v8h*)(ap);
    v8h a1 = *(const v8h*)(ap + 16);
    v8h b0 = *(const v8h*)(wp);
    v8h b1 = *(const v8h*)(wp + 16);
    v16h a = __builtin_shufflevector(a0, a1, 0,1,2,3,4,5,6,7,8,9,10,11,12,13,14,15);
    v16h b = __builtin_shufflevector(b0, b1, 0,1,2,3,4,5,6,7,8,9,10,11,12,13,14,15);
    acc = __builtin_amdgcn_wmma_f32_16x16x32_f16(false, a, false, b,
                                                 (short)0, acc, false, false);
    ap += 32; wp += 32;
  }

  const int ncol = ntile * 16 + lrow;
  if (ncol < N) {
    float bv = bias ? bias[ncol] : 0.f;
    const int mrow = mtile * 16 + lhi * 8;
    #pragma unroll
    for (int r = 0; r < 8; ++r) {
      float v = acc[r] + bv;
      if (act == 1) v = softplus_f(v);
      C[(size_t)(mrow + r) * ldc + ncol] = v;
    }
  }
}

// ---- depthwise causal conv (k=4, left pad 3) + SiLU, emits f32 + f16 ------
__global__ void k_dwconv_silu(const float* __restrict__ xr, int xrw, int di,
                              const float* __restrict__ cw, const float* __restrict__ cb,
                              float* __restrict__ xc, _Float16* __restrict__ xch,
                              int B, int L)
{
  int idx = blockIdx.x * blockDim.x + threadIdx.x;
  if (idx >= B * L * di) return;
  int c = idx % di; int row = idx / di;
  int l = row % L;  int b = row / L;
  float acc = cb[c];
  #pragma unroll
  for (int j = 0; j < 4; ++j) {
    int ls = l + j - 3;
    if (ls >= 0) acc += cw[c * 4 + j] * xr[(size_t)(b * L + ls) * xrw + c];
  }
  float v = silu_f(acc);
  xc[idx]  = v;
  xch[idx] = (_Float16)v;
}

// ---- selective scan: 16 lanes per channel (one per state n) ---------------
__global__ void k_scan(const float* __restrict__ delta, const float* __restrict__ xconv,
                       const float* __restrict__ xdbl, int ldxdbl, int dr,
                       const float* __restrict__ A_log, float* __restrict__ y,
                       int B, int L, int di)
{
  const int n    = threadIdx.x & 15;
  const int chan = blockIdx.x * 16 + (threadIdx.x >> 4);
  if (chan >= B * di) return;
  const int b = chan / di, d = chan % di;
  const float A = -__expf(A_log[d * 16 + n]);
  float h = 0.f;
  for (int l = 0; l < L; ++l) {
    const size_t row = (size_t)b * L + l;
    const float dt = delta[row * di + d];
    const float xt = xconv[row * di + d];
    const float bt = xdbl[row * ldxdbl + dr + n];
    const float ct = xdbl[row * ldxdbl + dr + 16 + n];
    h = h * __expf(dt * A) + (dt * xt) * bt;
    float yv = h * ct;
    #pragma unroll
    for (int off = 8; off > 0; off >>= 1) yv += __shfl_xor(yv, off, 16);
    if (n == 0) y[row * di + d] = yv;
  }
}

// ---- y2 = (yscan + xconv*D) * silu(res) -> f16 ---------------------------
__global__ void k_y2(const float* __restrict__ ys, const float* __restrict__ xc,
                     const float* __restrict__ xr, int xrw, int di,
                     const float* __restrict__ Dp, _Float16* __restrict__ y2h, int total)
{
  int idx = blockIdx.x * blockDim.x + threadIdx.x;
  if (idx >= total) return;
  int c = idx % di; size_t row = (size_t)idx / di;
  float res = xr[row * xrw + di + c];
  y2h[idx] = (_Float16)((ys[idx] + xc[idx] * Dp[c]) * silu_f(res));
}

// ---- (B*L, dm) -> NCHW with optional residual add -------------------------
__global__ void k_out_add(const float* __restrict__ mout, int dm,
                          const float* __restrict__ xpre, float* __restrict__ xout,
                          int B, int L, int add)
{
  int idx = blockIdx.x * blockDim.x + threadIdx.x;
  if (idx >= B * dm * L) return;
  int l = idx % L; int c = (idx / L) % dm; int b = idx / (L * dm);
  float v = mout[((size_t)b * L + l) * dm + c];
  if (add) v += xpre[idx];
  xout[idx] = v;
}

// ---- direct 3x3 conv, pad 1 ----------------------------------------------
__global__ void k_conv3x3(const float* __restrict__ in, const float* __restrict__ w,
                          const float* __restrict__ bias, float* __restrict__ out,
                          int B, int Cin, int Cout, int H, int W)
{
  int idx = blockIdx.x * blockDim.x + threadIdx.x;
  if (idx >= B * Cout * H * W) return;
  int x = idx % W; int y = (idx / W) % H;
  int co = (idx / (W * H)) % Cout; int b = idx / (W * H * Cout);
  float acc = bias[co];
  const float* wp = w + (size_t)co * Cin * 9;
  for (int ci = 0; ci < Cin; ++ci) {
    const float* ip = in + ((size_t)(b * Cin + ci) * H) * W;
    #pragma unroll
    for (int dy = 0; dy < 3; ++dy) {
      int yy = y + dy - 1; if ((unsigned)yy >= (unsigned)H) continue;
      #pragma unroll
      for (int dx = 0; dx < 3; ++dx) {
        int xx = x + dx - 1; if ((unsigned)xx >= (unsigned)W) continue;
        acc += wp[ci * 9 + dy * 3 + dx] * ip[yy * W + xx];
      }
    }
  }
  out[idx] = acc;
}

// ---- 1x1 conv -------------------------------------------------------------
__global__ void k_conv1x1(const float* __restrict__ in, const float* __restrict__ w,
                          const float* __restrict__ bias, float* __restrict__ out,
                          int B, int Cin, int Cout, int HW)
{
  int idx = blockIdx.x * blockDim.x + threadIdx.x;
  if (idx >= B * Cout * HW) return;
  int l = idx % HW; int co = (idx / HW) % Cout; int b = idx / (HW * Cout);
  float acc = bias[co];
  for (int ci = 0; ci < Cin; ++ci)
    acc += w[co * Cin + ci] * in[((size_t)(b * Cin + ci)) * HW + l];
  out[idx] = acc;
}

// ---- BN stats: one block per channel --------------------------------------
__global__ void k_bnstats(const float* __restrict__ x, int B, int C, int HW,
                          float* __restrict__ mean, float* __restrict__ rstd)
{
  __shared__ float s1[256], s2[256];
  const int c = blockIdx.x, N = B * HW;
  float a = 0.f, q = 0.f;
  for (int i = threadIdx.x; i < N; i += 256) {
    int b = i / HW, l = i % HW;
    float v = x[((size_t)(b * C + c)) * HW + l];
    a += v; q += v * v;
  }
  s1[threadIdx.x] = a; s2[threadIdx.x] = q; __syncthreads();
  for (int s = 128; s > 0; s >>= 1) {
    if (threadIdx.x < s) { s1[threadIdx.x] += s1[threadIdx.x + s]; s2[threadIdx.x] += s2[threadIdx.x + s]; }
    __syncthreads();
  }
  if (threadIdx.x == 0) {
    float m = s1[0] / N;
    mean[c] = m;
    rstd[c] = rsqrtf(s2[0] / N - m * m + 1e-5f);
  }
}

__global__ void k_bnrelu(const float* __restrict__ x, const float* __restrict__ mean,
                         const float* __restrict__ rstd, const float* __restrict__ g,
                         const float* __restrict__ bb, float* __restrict__ y,
                         int C, int HW, int total)
{
  int idx = blockIdx.x * blockDim.x + threadIdx.x;
  if (idx >= total) return;
  int c = (idx / HW) % C;
  float v = (x[idx] - mean[c]) * rstd[c] * g[c] + bb[c];
  y[idx] = v > 0.f ? v : 0.f;
}

__global__ void k_maxpool2(const float* __restrict__ in, float* __restrict__ out,
                           int B, int C, int H, int W)
{
  int OH = H / 2, OW = W / 2;
  int idx = blockIdx.x * blockDim.x + threadIdx.x;
  if (idx >= B * C * OH * OW) return;
  int ox = idx % OW; int oy = (idx / OW) % OH; int bc = idx / (OW * OH);
  const float* ip = in + ((size_t)bc * H + oy * 2) * W + ox * 2;
  float v = fmaxf(fmaxf(ip[0], ip[1]), fmaxf(ip[W], ip[W + 1]));
  out[idx] = v;
}

// ---- bilinear upsample x2 (align_corners) into concat buffer --------------
__global__ void k_up2_ac(const float* __restrict__ in, float* __restrict__ out,
                         int B, int C, int H, int W, int Ctot, int coff)
{
  int OH = 2 * H, OW = 2 * W;
  int idx = blockIdx.x * blockDim.x + threadIdx.x;
  if (idx >= B * C * OH * OW) return;
  int ox = idx % OW; int oy = (idx / OW) % OH;
  int c = (idx / (OW * OH)) % C; int b = idx / (OW * OH * C);
  float fy = oy * (float)(H - 1) / (float)(OH - 1);
  float fx = ox * (float)(W - 1) / (float)(OW - 1);
  int y0 = (int)floorf(fy), x0 = (int)floorf(fx);
  int y1 = min(y0 + 1, H - 1), x1 = min(x0 + 1, W - 1);
  float wy = fy - y0, wx = fx - x0;
  const float* ip = in + ((size_t)(b * C + c) * H) * W;
  float top = ip[y0 * W + x0] * (1.f - wx) + ip[y0 * W + x1] * wx;
  float bot = ip[y1 * W + x0] * (1.f - wx) + ip[y1 * W + x1] * wx;
  out[(((size_t)b * Ctot + coff + c) * OH + oy) * OW + ox] = top * (1.f - wy) + bot * wy;
}

__global__ void k_copych(const float* __restrict__ src, float* __restrict__ dst,
                         int B, int C, int HW, int Ctot, int coff)
{
  int idx = blockIdx.x * blockDim.x + threadIdx.x;
  if (idx >= B * C * HW) return;
  int l = idx % HW; int c = (idx / HW) % C; int b = idx / (HW * C);
  dst[((size_t)b * Ctot + coff + c) * HW + l] = src[idx];
}

// ---------------------------------------------------------------------------
struct MambaP {
  const float *W_in, *b_in, *conv_w, *conv_b, *Wx, *Wdt, *b_dt, *A_log, *D, *Wout, *b_out;
};

static void run_mamba(const MambaP& p, int dm, int dr, int B, int L,
                      const float* x_pre, float* x_out, int add,
                      float* ws, hipStream_t stream)
{
  const int di = 2 * dm, xrw = 4 * dm, M = B * L;
  const int ldx = dr + 32;                 // x_dbl width (34 / 36)
  const int nWx = DIV_UP(ldx, 16) * 16;    // 48

  _Float16* seqh  = (_Float16*)(ws + OFF_SEQH);
  float*    xr    = ws + OFF_BIG;
  float*    xconv = ws + OFF_XCONV;
  _Float16* xch   = (_Float16*)(ws + OFF_XCONVH);
  float*    xdbl  = ws + OFF_XDBL;
  _Float16* dtinh = (_Float16*)(ws + OFF_DTINH);
  float*    delta = ws + OFF_DELTA;
  float*    yscan = ws + OFF_YSCAN;
  _Float16* y2h   = (_Float16*)(ws + OFF_Y2H);
  float*    mout  = ws + OFF_MOUT;
  _Float16* wh    = (_Float16*)(ws + OFF_WH);

  // weight casts to padded f16
  k_cast_w16<<<DIV_UP(2*di*dm,256),256,0,stream>>>(p.W_in, 2*di, dm, wh+WH_WIN, 2*di, dm);
  k_cast_w16<<<DIV_UP(nWx*di,256),256,0,stream>>>(p.Wx,  ldx,  di, wh+WH_WX,  nWx,  di);
  k_cast_w16<<<DIV_UP(di*32,256),256,0,stream>>>(p.Wdt,  di,   dr, wh+WH_WDT, di,   32);
  k_cast_w16<<<DIV_UP(dm*di,256),256,0,stream>>>(p.Wout, dm,   di, wh+WH_WOUT,dm,   di);

  // NCHW -> seq f16
  k_nchw_to_seq<<<DIV_UP(B*dm*L,256),256,0,stream>>>(x_pre, seqh, B, dm, L);

  // xr = seq @ W_in^T + b_in       (M x 4dm)
  { dim3 g(M/64, (2*di)/16);
    k_gemm_wmma<<<g,128,0,stream>>>(seqh, dm, wh+WH_WIN, dm, p.b_in, xr, 2*di, 2*di, dm/32, 0); }

  // depthwise causal conv + silu
  k_dwconv_silu<<<DIV_UP(M*di,256),256,0,stream>>>(xr, xrw, di, p.conv_w, p.conv_b, xconv, xch, B, L);

  // x_dbl = xc @ Wx^T              (M x (dr+32))
  { dim3 g(M/64, nWx/16);
    k_gemm_wmma<<<g,128,0,stream>>>(xch, di, wh+WH_WX, di, nullptr, xdbl, ldx, ldx, di/32, 0); }

  // delta input slice -> padded f16
  k_cast_a16<<<DIV_UP(M*32,256),256,0,stream>>>(xdbl, ldx, dr, dtinh, 32, M);

  // delta = softplus(din @ Wdt^T + b_dt)   (M x di)
  { dim3 g(M/64, di/16);
    k_gemm_wmma<<<g,128,0,stream>>>(dtinh, 32, wh+WH_WDT, 32, p.b_dt, delta, di, di, 1, 1); }

  // selective scan
  k_scan<<<DIV_UP(B*di,16),256,0,stream>>>(delta, xconv, xdbl, ldx, dr, p.A_log, yscan, B, L, di);

  // gate: (y + xc*D) * silu(res) -> f16
  k_y2<<<DIV_UP(M*di,256),256,0,stream>>>(yscan, xconv, xr, xrw, di, p.D, y2h, M*di);

  // out = y2 @ Wout^T + b_out      (M x dm)
  { dim3 g(M/64, dm/16);
    k_gemm_wmma<<<g,128,0,stream>>>(y2h, di, wh+WH_WOUT, di, p.b_out, mout, dm, dm, di/32, 0); }

  // back to NCHW (+residual)
  k_out_add<<<DIV_UP(B*dm*L,256),256,0,stream>>>(mout, dm, x_pre, x_out, B, L, add);
}

// ---------------------------------------------------------------------------
extern "C" void kernel_launch(void* const* d_in, const int* in_sizes, int n_in,
                              void* d_out, int out_size, void* d_ws, size_t ws_size,
                              hipStream_t stream)
{
  auto F = [&](int i) { return (const float*)d_in[i]; };
  float* ws = (float*)d_ws;
  const int B = 4, H = 128, W = 128, HW = H * W;

  int i = 1;
  const float *e1c1_w=F(i++), *e1c1_b=F(i++), *e1bn1_g=F(i++), *e1bn1_b=F(i++);
  const float *e1c2_w=F(i++), *e1c2_b=F(i++), *e1bn2_g=F(i++), *e1bn2_b=F(i++);
  MambaP m1; m1.W_in=F(i++); m1.b_in=F(i++); m1.conv_w=F(i++); m1.conv_b=F(i++);
  m1.Wx=F(i++); m1.Wdt=F(i++); m1.b_dt=F(i++); m1.A_log=F(i++); m1.D=F(i++);
  m1.Wout=F(i++); m1.b_out=F(i++);
  const float *e2c1_w=F(i++), *e2c1_b=F(i++), *e2bn1_g=F(i++), *e2bn1_b=F(i++);
  MambaP m2; m2.W_in=F(i++); m2.b_in=F(i++); m2.conv_w=F(i++); m2.conv_b=F(i++);
  m2.Wx=F(i++); m2.Wdt=F(i++); m2.b_dt=F(i++); m2.A_log=F(i++); m2.D=F(i++);
  m2.Wout=F(i++); m2.b_out=F(i++);
  MambaP mb; mb.W_in=F(i++); mb.b_in=F(i++); mb.conv_w=F(i++); mb.conv_b=F(i++);
  mb.Wx=F(i++); mb.Wdt=F(i++); mb.b_dt=F(i++); mb.A_log=F(i++); mb.D=F(i++);
  mb.Wout=F(i++); mb.b_out=F(i++);
  const float *d1c1_w=F(i++), *d1c1_b=F(i++), *d1bn1_g=F(i++), *d1bn1_b=F(i++);
  const float *d1c2_w=F(i++), *d1c2_b=F(i++), *d1bn2_g=F(i++), *d1bn2_b=F(i++);
  const float *fin_w=F(i++), *fin_b=F(i++);

  float* mean = ws + OFF_STATS;
  float* rstd = ws + OFF_STATS + 128;

  // -------- encoder stage 1 (128x128, 32 ch) --------
  k_conv3x3<<<DIV_UP(B*32*HW,256),256,0,stream>>>(F(0), e1c1_w, e1c1_b, ws+OFF_BUFA, B, 2, 32, H, W);
  k_bnstats<<<32,256,0,stream>>>(ws+OFF_BUFA, B, 32, HW, mean, rstd);
  k_bnrelu<<<DIV_UP(B*32*HW,256),256,0,stream>>>(ws+OFF_BUFA, mean, rstd, e1bn1_g, e1bn1_b, ws+OFF_BUFB, 32, HW, B*32*HW);
  k_conv3x3<<<DIV_UP(B*32*HW,256),256,0,stream>>>(ws+OFF_BUFB, e1c2_w, e1c2_b, ws+OFF_BUFA, B, 32, 32, H, W);
  k_bnstats<<<32,256,0,stream>>>(ws+OFF_BUFA, B, 32, HW, mean, rstd);
  k_bnrelu<<<DIV_UP(B*32*HW,256),256,0,stream>>>(ws+OFF_BUFA, mean, rstd, e1bn2_g, e1bn2_b, ws+OFF_BUFB, 32, HW, B*32*HW);

  // x1 = x1 + mamba2d(x1)
  run_mamba(m1, 32, 2, B, HW, ws+OFF_BUFB, ws+OFF_X1, 1, ws, stream);

  // -------- encoder stage 2 (64x64, 64 ch) --------
  k_maxpool2<<<DIV_UP(B*32*(HW/4),256),256,0,stream>>>(ws+OFF_X1, ws+OFF_P1, B, 32, H, W);
  const int H2 = 64, W2 = 64, HW2 = H2 * W2;
  k_conv3x3<<<DIV_UP(B*64*HW2,256),256,0,stream>>>(ws+OFF_P1, e2c1_w, e2c1_b, ws+OFF_BUFA, B, 32, 64, H2, W2);
  k_bnstats<<<64,256,0,stream>>>(ws+OFF_BUFA, B, 64, HW2, mean, rstd);
  k_bnrelu<<<DIV_UP(B*64*HW2,256),256,0,stream>>>(ws+OFF_BUFA, mean, rstd, e2bn1_g, e2bn1_b, ws+OFF_BUFB, 64, HW2, B*64*HW2);

  run_mamba(m2, 64, 4, B, HW2, ws+OFF_BUFB, ws+OFF_X2, 1, ws, stream);   // x2
  run_mamba(mb, 64, 4, B, HW2, ws+OFF_X2,  ws+OFF_XB, 0, ws, stream);    // bottleneck

  // -------- decoder --------
  // concat buffer: ch 0..63 = upsample(xb), ch 64..95 = x1
  k_up2_ac<<<DIV_UP(B*64*HW,256),256,0,stream>>>(ws+OFF_XB, ws+OFF_BIG, B, 64, H2, W2, 96, 0);
  k_copych<<<DIV_UP(B*32*HW,256),256,0,stream>>>(ws+OFF_X1, ws+OFF_BIG, B, 32, HW, 96, 64);

  k_conv3x3<<<DIV_UP(B*32*HW,256),256,0,stream>>>(ws+OFF_BIG, d1c1_w, d1c1_b, ws+OFF_BUFA, B, 96, 32, H, W);
  k_bnstats<<<32,256,0,stream>>>(ws+OFF_BUFA, B, 32, HW, mean, rstd);
  k_bnrelu<<<DIV_UP(B*32*HW,256),256,0,stream>>>(ws+OFF_BUFA, mean, rstd, d1bn1_g, d1bn1_b, ws+OFF_BUFB, 32, HW, B*32*HW);
  k_conv3x3<<<DIV_UP(B*32*HW,256),256,0,stream>>>(ws+OFF_BUFB, d1c2_w, d1c2_b, ws+OFF_BUFA, B, 32, 32, H, W);
  k_bnstats<<<32,256,0,stream>>>(ws+OFF_BUFA, B, 32, HW, mean, rstd);
  k_bnrelu<<<DIV_UP(B*32*HW,256),256,0,stream>>>(ws+OFF_BUFA, mean, rstd, d1bn2_g, d1bn2_b, ws+OFF_BUFB, 32, HW, B*32*HW);

  // final 1x1 conv -> out (B,2,128,128)
  k_conv1x1<<<DIV_UP(B*2*HW,256),256,0,stream>>>(ws+OFF_BUFB, fin_w, fin_b, (float*)d_out, B, 32, 2, HW);
}
